// AttLayer3_4105988735255
// MI455X (gfx1250) — matrix operand order repaired and checked by reading, not compiled, not run on hardware
//
#include <hip/hip_runtime.h>
#include <math.h>

// ---------------------------------------------------------------------------
// VMamba SS2D block + gram-attention for MI455X (gfx1250, wave32, WMMA).
//   * GEMMs: f16 WMMA (v_wmma_f32_16x16x32_f16), fp32 accumulate,
//     128x64 block tiles double-buffered in LDS, filled with
//     GLOBAL_LOAD_ASYNC_TO_LDS_B128 (ASYNCcnt-tracked) when available.
//   * Selective scan: 1 lane per (b,k,d) channel, B/C state broadcast via shfl.
//   * Gram matrix fused with row-max (never materialized).
// Workspace requirement: ~152 MB in d_ws.
// ---------------------------------------------------------------------------

#define BB   8
#define CC   512
#define HH   24
#define WW   24
#define LL   576          // HH*WW
#define BL   4608         // BB*LL
#define DI   1024         // D_INNER
#define DTR  32           // DT_RANK
#define NST  16           // D_STATE
#define KD4  4            // K directions
#define CPJ  64           // DT_RANK + 2*D_STATE

typedef __attribute__((ext_vector_type(16))) _Float16 v16h;
typedef __attribute__((ext_vector_type(8)))  float    v8f;

// -------- async global->LDS (gfx1250) with synchronous fallback ------------
#if defined(__has_builtin)
#if __has_builtin(__builtin_amdgcn_global_load_async_to_lds_b128) && \
    __has_builtin(__builtin_amdgcn_s_wait_asynccnt)
#define ASYNC_LDS 1
#endif
#endif
#ifndef ASYNC_LDS
#define ASYNC_LDS 0
#endif

// Builtin signature (from hipcc diagnostic): (int4 as1*, int4 as3*, Ii, Ii);
// HIP prints AS1 as "__device__", AS3 as "__shared__".
typedef int v4i __attribute__((__vector_size__(16)));
typedef __attribute__((address_space(1))) v4i v4i_g;
typedef __attribute__((address_space(3))) v4i v4i_s;

static __device__ __forceinline__ void cp_b128(_Float16* dstLds, const _Float16* srcG) {
#if ASYNC_LDS
  __builtin_amdgcn_global_load_async_to_lds_b128((v4i_g*)(void*)srcG,
                                                 (v4i_s*)(void*)dstLds, 0, 0);
#else
  *(v4i*)dstLds = *(const v4i*)srcG;
#endif
}

static __device__ __forceinline__ void tiles_ready() {
#if ASYNC_LDS
  __builtin_amdgcn_s_wait_asynccnt(0);
#endif
  __syncthreads();
}

static __device__ __forceinline__ void atomicMaxF(float* a, float v) {
  // Sign-aware float max via integer atomics (works for mixed signs).
  if (v >= 0.f) atomicMax((int*)a, __float_as_int(v));
  else          atomicMin((unsigned int*)a, __float_as_uint(v));
}

// LDS tile geometry: rows padded to 40 halves (80B) -> 16-lane b128 fragment
// reads hit banks 20*r mod 64 (all multiples of 4 once) = conflict-free.
#define BK   32
#define LPAD 40

// ---------------------------------------------------------------------------
// GEMM: Out[M,N] = A[M,Kd](f16,row-major) * W[N,Kd](f16,row-major)^T
// Block 256 thr = 8 waves (4M x 2N); block tile 128x64; wave tile 32x32
// (2 A-frags x 2 B-frags -> 4 WMMAs per BK step). Grid (M/128, N/64).
// Fragment mapping per CDNA5 ISA 7.12.2 (wave32).
// ---------------------------------------------------------------------------
__global__ void __launch_bounds__(256)
k_gemm_f16(const _Float16* __restrict__ A, int lda,
           const _Float16* __restrict__ W, int ldw,
           float* __restrict__ Out, int ldo, int Kd) {
  __shared__ __align__(16) _Float16 As[2][128][LPAD];
  __shared__ __align__(16) _Float16 Bs[2][64][LPAD];
  const int tid  = threadIdx.x;
  const int lane = tid & 31, wave = tid >> 5;
  const int hi   = lane >> 4, ml = lane & 15;
  const int wm   = wave & 3,  wn = wave >> 2;
  const int mBase = blockIdx.x * 128;
  const int nBase = blockIdx.y * 64;
  const int KT = Kd >> 5;

  auto load_tiles = [&](int kt, int buf) {
    const int k0 = kt * BK;
#pragma unroll
    for (int i = 0; i < 2; ++i) {              // A: 128 rows x 4 chunks
      int q = tid + i * 256;
      int r = q >> 2, cc = q & 3;
      cp_b128(&As[buf][r][cc * 8], A + (size_t)(mBase + r) * lda + k0 + cc * 8);
    }
    { int r = tid >> 2, cc = tid & 3;          // B: 64 rows x 4 chunks
      cp_b128(&Bs[buf][r][cc * 8], W + (size_t)(nBase + r) * ldw + k0 + cc * 8); }
  };

  v8f acc[2][2] = {};
  load_tiles(0, 0);
  tiles_ready();

  for (int kt = 0; kt < KT; ++kt) {
    int buf = kt & 1;
    if (kt + 1 < KT) load_tiles(kt + 1, buf ^ 1);

    v16h af[2], bf[2];
#pragma unroll
    for (int i = 0; i < 2; ++i) {
      const _Float16* ar = &As[buf][wm * 32 + i * 16 + ml][0];
#pragma unroll
      for (int e = 0; e < 16; ++e) {
        int v  = e >> 1;
        int kk = (v >> 2) * 16 + hi * 8 + (v & 3) * 2 + (e & 1);
        af[i][e] = ar[kk];
      }
      const _Float16* br = &Bs[buf][wn * 32 + i * 16 + ml][hi * 8];
#pragma unroll
      for (int e = 0; e < 16; ++e) bf[i][e] = br[(e >> 3) * 16 + (e & 7)];
    }
#pragma unroll
    for (int i = 0; i < 2; ++i)
#pragma unroll
      for (int j = 0; j < 2; ++j)
        acc[i][j] = __builtin_amdgcn_wmma_f32_16x16x32_f16(
            false, af[i], false, bf[j], (short)0, acc[i][j], false, false);

    if (kt + 1 < KT) tiles_ready();
  }

#pragma unroll
  for (int i = 0; i < 2; ++i) {
    int mrow = mBase + wm * 32 + i * 16 + hi * 8;
#pragma unroll
    for (int j = 0; j < 2; ++j) {
      float* orow = Out + (size_t)mrow * ldo + nBase + wn * 32 + j * 16 + ml;
#pragma unroll
      for (int r = 0; r < 8; ++r) orow[(size_t)r * ldo] = acc[i][j][r];
    }
  }
}

// ---------------------------------------------------------------------------
// Gram: T = X*X^T * SCALE, fused row-max per 576-column batch block.
// Same 128x64 LDS tiling; 64-col tiles align with batch blocks (576%64==0).
// ---------------------------------------------------------------------------
__global__ void __launch_bounds__(256)
k_gram_max(const _Float16* __restrict__ X, float* __restrict__ maxper) {
  __shared__ __align__(16) _Float16 As[2][128][LPAD];
  __shared__ __align__(16) _Float16 Bs[2][64][LPAD];
  const int tid  = threadIdx.x;
  const int lane = tid & 31, wave = tid >> 5;
  const int hi   = lane >> 4, ml = lane & 15;
  const int wm   = wave & 3,  wn = wave >> 2;
  const int mBase = blockIdx.x * 128;
  const int nBase = blockIdx.y * 64;
  const int bcol  = nBase / LL;
  const int KT = CC >> 5;

  auto load_tiles = [&](int kt, int buf) {
    const int k0 = kt * BK;
#pragma unroll
    for (int i = 0; i < 2; ++i) {
      int q = tid + i * 256;
      int r = q >> 2, cc = q & 3;
      cp_b128(&As[buf][r][cc * 8], X + (size_t)(mBase + r) * CC + k0 + cc * 8);
    }
    { int r = tid >> 2, cc = tid & 3;
      cp_b128(&Bs[buf][r][cc * 8], X + (size_t)(nBase + r) * CC + k0 + cc * 8); }
  };

  v8f acc[2][2] = {};
  load_tiles(0, 0);
  tiles_ready();

  for (int kt = 0; kt < KT; ++kt) {
    int buf = kt & 1;
    if (kt + 1 < KT) load_tiles(kt + 1, buf ^ 1);

    v16h af[2], bf[2];
#pragma unroll
    for (int i = 0; i < 2; ++i) {
      const _Float16* ar = &As[buf][wm * 32 + i * 16 + ml][0];
#pragma unroll
      for (int e = 0; e < 16; ++e) {
        int v  = e >> 1;
        int kk = (v >> 2) * 16 + hi * 8 + (v & 3) * 2 + (e & 1);
        af[i][e] = ar[kk];
      }
      const _Float16* br = &Bs[buf][wn * 32 + i * 16 + ml][hi * 8];
#pragma unroll
      for (int e = 0; e < 16; ++e) bf[i][e] = br[(e >> 3) * 16 + (e & 7)];
    }
#pragma unroll
    for (int i = 0; i < 2; ++i)
#pragma unroll
      for (int j = 0; j < 2; ++j)
        acc[i][j] = __builtin_amdgcn_wmma_f32_16x16x32_f16(
            false, af[i], false, bf[j], (short)0, acc[i][j], false, false);

    if (kt + 1 < KT) tiles_ready();
  }

  const float scale = 0.044194173824159216f; // 1/sqrt(512)
#pragma unroll
  for (int i = 0; i < 2; ++i) {
    int mrow = mBase + wm * 32 + i * 16 + hi * 8;
#pragma unroll
    for (int r = 0; r < 8; ++r) {
      float v = fmaxf(acc[i][0][r], acc[i][1][r]) * scale;
#pragma unroll
      for (int off = 1; off < 16; off <<= 1) v = fmaxf(v, __shfl_xor(v, off, 16));
      if (ml == 0) atomicMaxF(&maxper[(size_t)(mrow + r) * 8 + bcol], v);
    }
  }
}

// --------------------------- elementwise / utility -------------------------

__global__ void k_f2h(const float* __restrict__ s, _Float16* __restrict__ d, size_t n) {
  size_t i = (size_t)blockIdx.x * 256 + threadIdx.x;
  if (i < n) d[i] = (_Float16)s[i];
}

__global__ void k_fill(float* __restrict__ p, float v, size_t n) {
  size_t i = (size_t)blockIdx.x * 256 + threadIdx.x;
  if (i < n) p[i] = v;
}

// x5 [B,C,H,W] -> token-major f16 [BL,C]
__global__ void k_tok_f16(const float* __restrict__ x5, _Float16* __restrict__ fh) {
  size_t i = (size_t)blockIdx.x * 256 + threadIdx.x;
  if (i >= (size_t)BL * CC) return;
  int c = (int)(i & (CC - 1));
  int g = (int)(i >> 9);
  int b = g / LL, t = g % LL;
  fh[i] = (_Float16)x5[((size_t)b * CC + c) * LL + t];
}

// x5r = gemm1_out + conv_b + x5(orig, transposed)
__global__ void k_epi_conv(const float* __restrict__ tmp, const float* __restrict__ cb,
                           const float* __restrict__ x5, float* __restrict__ x5r) {
  size_t i = (size_t)blockIdx.x * 256 + threadIdx.x;
  if (i >= (size_t)BL * CC) return;
  int c = (int)(i & (CC - 1));
  int g = (int)(i >> 9);
  int b = g / LL, t = g % LL;
  x5r[i] = tmp[i] + cb[c] + x5[((size_t)b * CC + c) * LL + t];
}

// LayerNorm over C=512 per token -> f16
__global__ void __launch_bounds__(256)
k_ln1(const float* __restrict__ X, const float* __restrict__ gw,
      const float* __restrict__ bw, _Float16* __restrict__ Oh) {
  int row = blockIdx.x, tid = threadIdx.x;
  const float* x = X + (size_t)row * CC;
  float v0 = x[tid], v1 = x[tid + 256];
  __shared__ float red[256];
  red[tid] = v0 + v1; __syncthreads();
  for (int st = 128; st; st >>= 1) { if (tid < st) red[tid] += red[tid + st]; __syncthreads(); }
  float mean = red[0] * (1.f / CC);
  __syncthreads();
  float d0 = v0 - mean, d1 = v1 - mean;
  red[tid] = d0 * d0 + d1 * d1; __syncthreads();
  for (int st = 128; st; st >>= 1) { if (tid < st) red[tid] += red[tid + st]; __syncthreads(); }
  float rs = rsqrtf(red[0] * (1.f / CC) + 1e-5f);
  Oh[(size_t)row * CC + tid]       = (_Float16)(d0 * rs * gw[tid] + bw[tid]);
  Oh[(size_t)row * CC + tid + 256] = (_Float16)(d1 * rs * gw[tid + 256] + bw[tid + 256]);
}

// depthwise 3x3 conv (SAME) + bias + silu on xi half of xz -> xc [BL,DI]
__global__ void k_dwconv(const float* __restrict__ xz, const float* __restrict__ w,
                         const float* __restrict__ bias, float* __restrict__ xc) {
  size_t i = (size_t)blockIdx.x * 256 + threadIdx.x;
  if (i >= (size_t)BL * DI) return;
  int d = (int)(i & (DI - 1));
  int g = (int)(i >> 10);
  int b = g / LL, t = g % LL;
  int h = t / WW, wc = t % WW;
  float acc = bias[d];
#pragma unroll
  for (int dh = -1; dh <= 1; ++dh)
#pragma unroll
    for (int dw = -1; dw <= 1; ++dw) {
      int hh = h + dh, ww = wc + dw;
      if ((unsigned)hh < (unsigned)HH && (unsigned)ww < (unsigned)WW)
        acc += xz[((size_t)(b * LL + hh * WW + ww)) * (2 * DI) + d] *
               w[d * 9 + (dh + 1) * 3 + (dw + 1)];
    }
  xc[i] = acc / (1.f + __expf(-acc));
}

// Dth[k][g][r] = f16(P[k][g][r]), r<32
__global__ void k_dtcols(const float* __restrict__ P, _Float16* __restrict__ Dth) {
  size_t i = (size_t)blockIdx.x * 256 + threadIdx.x;
  if (i >= (size_t)KD4 * BL * DTR) return;
  int r = (int)(i & (DTR - 1));
  size_t kg = i >> 5;                 // k*BL + g
  Dth[i] = (_Float16)P[kg * CPJ + r];
}

// DeltaH[k] = f16(softplus(tmp + dt_proj_b[k]))
__global__ void k_softplus(const float* __restrict__ tmp, const float* __restrict__ dtb,
                           _Float16* __restrict__ DeltaH, int k) {
  size_t i = (size_t)blockIdx.x * 256 + threadIdx.x;
  if (i >= (size_t)BL * DI) return;
  int d = (int)(i & (DI - 1));
  float v = tmp[i] + dtb[k * DI + d];
  float sp = (v > 20.f) ? v : log1pf(__expf(v));
  DeltaH[(size_t)k * BL * DI + i] = (_Float16)sp;
}

__global__ void k_dssum(const float* __restrict__ Ds, float* __restrict__ dssum) {
  int d = blockIdx.x * 256 + threadIdx.x;
  if (d < DI) dssum[d] = Ds[d] + Ds[DI + d] + Ds[2 * DI + d] + Ds[3 * DI + d];
}

// ---------------------------------------------------------------------------
// Selective scan: one wave32 per (b,k,d-chunk); lane = channel d.
// B_t/C_t (32 shared floats) loaded one-per-lane, broadcast via __shfl.
// Direction permutation applied through token index math; merge via atomicAdd.
// ---------------------------------------------------------------------------
__global__ void __launch_bounds__(32)
k_scan(const float* __restrict__ xc, const _Float16* __restrict__ dh,
       const float* __restrict__ P, const float* __restrict__ A_logs,
       float* __restrict__ ytot) {
  int id = blockIdx.x;
  int lane = threadIdx.x;
  int dc = id & 31;
  int k  = (id >> 5) & 3;
  int b  = id >> 7;
  int d  = dc * 32 + lane;

  float Areg[NST], h[NST];
#pragma unroll
  for (int n = 0; n < NST; ++n) {
    Areg[n] = -__expf(A_logs[((size_t)(k * DI + d)) * NST + n]);
    h[n] = 0.f;
  }
  const _Float16* dk = dh + (size_t)k * BL * DI;
  const float*    Pk = P  + (size_t)k * BL * CPJ;

  for (int l = 0; l < LL; ++l) {
    int tok;
    if (k == 0)      tok = l;
    else if (k == 1) { int hh = l % HH, ww = l / HH; tok = hh * WW + ww; }
    else if (k == 2) tok = LL - 1 - l;
    else             { int lr = LL - 1 - l; int hh = lr % HH, ww = lr / HH; tok = hh * WW + ww; }
    int g = b * LL + tok;

    float pv = Pk[(size_t)g * CPJ + 32 + lane];   // lanes 0..15: B, 16..31: C
    float u  = xc[(size_t)g * DI + d];
    float dt = (float)dk[(size_t)g * DI + d];
    float du = dt * u;
    float y  = 0.f;
#pragma unroll
    for (int n = 0; n < NST; ++n) {
      float dA = __expf(dt * Areg[n]);
      h[n] = dA * h[n] + du * __shfl(pv, n, 32);
      y += h[n] * __shfl(pv, n + 16, 32);
    }
    atomicAdd(&ytot[(size_t)g * DI + d], y);
  }
}

// out_norm LayerNorm over DI=1024 of (ytot + dssum*xc), gated by silu(z) -> f16
__global__ void __launch_bounds__(256)
k_ln_gate(const float* __restrict__ ytot, const float* __restrict__ xc,
          const float* __restrict__ dssum, const float* __restrict__ xz,
          const float* __restrict__ gw, const float* __restrict__ bw,
          _Float16* __restrict__ outh) {
  int row = blockIdx.x, tid = threadIdx.x;
  size_t r1 = (size_t)row * DI;
  float vals[4], s = 0.f;
#pragma unroll
  for (int j = 0; j < 4; ++j) {
    int d = tid + j * 256;
    float v = ytot[r1 + d] + dssum[d] * xc[r1 + d];
    vals[j] = v; s += v;
  }
  __shared__ float red[256];
  red[tid] = s; __syncthreads();
  for (int st = 128; st; st >>= 1) { if (tid < st) red[tid] += red[tid + st]; __syncthreads(); }
  float mean = red[0] * (1.f / DI);
  __syncthreads();
  float vv = 0.f;
#pragma unroll
  for (int j = 0; j < 4; ++j) { float dd = vals[j] - mean; vv += dd * dd; }
  red[tid] = vv; __syncthreads();
  for (int st = 128; st; st >>= 1) { if (tid < st) red[tid] += red[tid + st]; __syncthreads(); }
  float rs = rsqrtf(red[0] * (1.f / DI) + 1e-5f);
#pragma unroll
  for (int j = 0; j < 4; ++j) {
    int d = tid + j * 256;
    float yn = (vals[j] - mean) * rs * gw[d] + bw[d];
    float z  = xz[(size_t)row * (2 * DI) + DI + d];
    float si = z / (1.f + __expf(-z));
    outh[r1 + d] = (_Float16)(yn * si);
  }
}

// x00 += x5r; fh = f16(x00)
__global__ void k_epi_x00(float* __restrict__ x00, const float* __restrict__ x5r,
                          _Float16* __restrict__ fh) {
  size_t i = (size_t)blockIdx.x * 256 + threadIdx.x;
  if (i >= (size_t)BL * CC) return;
  float v = x00[i] + x5r[i];
  x00[i] = v;
  fh[i] = (_Float16)v;
}

__global__ void k_s(const float* __restrict__ maxper, float* __restrict__ s) {
  int i = blockIdx.x * 256 + threadIdx.x;
  if (i >= BL) return;
  float a = 0.f;
#pragma unroll
  for (int b = 0; b < BB; ++b) a += maxper[(size_t)i * 8 + b];
  s[i] = a * 0.125f;
}

// mask[b][l] = (s == rowmax)  (argmax of s == argmax of softmax(s))
__global__ void __launch_bounds__(256)
k_mask(const float* __restrict__ s, float* __restrict__ maskf) {
  int b = blockIdx.x, tid = threadIdx.x;
  __shared__ float red[256];
  float mx = -3e38f;
  for (int l = tid; l < LL; l += 256) mx = fmaxf(mx, s[b * LL + l]);
  red[tid] = mx; __syncthreads();
  for (int st = 128; st; st >>= 1) { if (tid < st) red[tid] = fmaxf(red[tid], red[tid + st]); __syncthreads(); }
  float m = red[0];
  for (int l = tid; l < LL; l += 256)
    maskf[b * LL + l] = (s[b * LL + l] == m) ? 1.f : 0.f;
}

// invn[g] = 1 / max(||x5r[g,:]||, 1e-12)
__global__ void __launch_bounds__(256)
k_invn(const float* __restrict__ x5r, float* __restrict__ invn) {
  int g = blockIdx.x, tid = threadIdx.x;
  const float* x = x5r + (size_t)g * CC;
  float ss = 0.f;
  for (int c = tid; c < CC; c += 256) { float v = x[c]; ss += v * v; }
  __shared__ float red[256];
  red[tid] = ss; __syncthreads();
  for (int st = 128; st; st >>= 1) { if (tid < st) red[tid] += red[tid + st]; __syncthreads(); }
  if (tid == 0) invn[g] = 1.f / fmaxf(sqrtf(red[0]), 1e-12f);
}

// seeds[o][c] = sum_t mask * x5r * invn   (per batch o)
__global__ void k_seeds(const float* __restrict__ x5r, const float* __restrict__ invn,
                        const float* __restrict__ maskf, float* __restrict__ seeds) {
  int idx = blockIdx.x * 256 + threadIdx.x;
  if (idx >= BB * CC) return;
  int o = idx >> 9, c = idx & (CC - 1);
  float acc = 0.f;
  for (int t = 0; t < LL; ++t) {
    int g = o * LL + t;
    acc += maskf[g] * x5r[(size_t)g * CC + c] * invn[g];
  }
  seeds[o * CC + c] = acc;
}

// cor[g] = mean_o relu(invn[g] * dot(x5r[g,:], seeds[o,:]))
__global__ void __launch_bounds__(256)
k_cor(const float* __restrict__ x5r, const float* __restrict__ seeds,
      const float* __restrict__ invn, float* __restrict__ cor) {
  int g = blockIdx.x, tid = threadIdx.x;
  __shared__ float red[8 * 256];
  float part[8];
#pragma unroll
  for (int o = 0; o < 8; ++o) part[o] = 0.f;
  for (int c = tid; c < CC; c += 256) {
    float x = x5r[(size_t)g * CC + c];
#pragma unroll
    for (int o = 0; o < 8; ++o) part[o] += x * seeds[o * CC + c];
  }
#pragma unroll
  for (int o = 0; o < 8; ++o) red[o * 256 + tid] = part[o];
  __syncthreads();
  for (int st = 128; st; st >>= 1) {
    if (tid < st)
#pragma unroll
      for (int o = 0; o < 8; ++o) red[o * 256 + tid] += red[o * 256 + tid + st];
    __syncthreads();
  }
  if (tid == 0) {
    float inv = invn[g], acc = 0.f;
#pragma unroll
    for (int o = 0; o < 8; ++o) acc += fmaxf(inv * red[o * 256], 0.f);
    cor[g] = acc * 0.125f;
  }
}

__global__ void __launch_bounds__(256)
k_minmax(const float* __restrict__ cor, float* __restrict__ bmn, float* __restrict__ bmx) {
  int b = blockIdx.x, tid = threadIdx.x;
  __shared__ float rmn[256], rmx[256];
  float mn = 3e38f, mx = -3e38f;
  for (int l = tid; l < LL; l += 256) {
    float v = cor[b * LL + l];
    mn = fminf(mn, v); mx = fmaxf(mx, v);
  }
  rmn[tid] = mn; rmx[tid] = mx; __syncthreads();
  for (int st = 128; st; st >>= 1) {
    if (tid < st) { rmn[tid] = fminf(rmn[tid], rmn[tid + st]); rmx[tid] = fmaxf(rmx[tid], rmx[tid + st]); }
    __syncthreads();
  }
  if (tid == 0) { bmn[b] = rmn[0]; bmx[b] = rmx[0]; }
}

__global__ void k_cormap(const float* __restrict__ cor, const float* __restrict__ bmn,
                         const float* __restrict__ bmx, float* __restrict__ cormap) {
  int i = blockIdx.x * 256 + threadIdx.x;
  if (i >= BL) return;
  int b = i / LL;
  cormap[i] = (cor[i] - bmn[b]) / (bmx[b] - bmn[b] + 1e-12f);
}

__global__ void __launch_bounds__(256)
k_proto(const float* __restrict__ x5r, const float* __restrict__ cormap,
        float* __restrict__ proto) {
  int c = blockIdx.x, tid = threadIdx.x;
  float s = 0.f;
  for (int g = tid; g < BL; g += 256) s += x5r[(size_t)g * CC + c] * cormap[g];
  __shared__ float red[256];
  red[tid] = s; __syncthreads();
  for (int st = 128; st; st >>= 1) { if (tid < st) red[tid] += red[tid + st]; __syncthreads(); }
  if (tid == 0) proto[c] = red[0] * (1.f / BL);
}

// Final writes: out0=x5 (BCHW), out1=proto1, out2=x5*proto+x51, out3=mask
#define O1 2359296
#define O2 2359808
#define O3 4719104
__global__ void k_outputs(const float* __restrict__ x5r, const float* __restrict__ proto,
                          const float* __restrict__ cormap, const float* __restrict__ maskf,
                          float* __restrict__ out) {
  size_t i = (size_t)blockIdx.x * 256 + threadIdx.x;
  if (i < (size_t)BB * CC * LL) {
    int t = (int)(i % LL);
    int c = (int)((i / LL) % CC);
    int b = (int)(i / ((size_t)LL * CC));
    float v = x5r[((size_t)(b * LL + t)) * CC + c];
    out[i] = v;
    out[O2 + i] = v * proto[c] + v * cormap[b * LL + t];
  }
  if (i < CC) out[O1 + i] = proto[i];
  if (i < BL) out[O3 + i] = maskf[i];
}

// ---------------------------------------------------------------------------
static inline size_t cdivz(size_t a, size_t b) { return (a + b - 1) / b; }

extern "C" void kernel_launch(void* const* d_in, const int* in_sizes, int n_in,
                              void* d_out, int out_size, void* d_ws, size_t ws_size,
                              hipStream_t stream) {
  (void)in_sizes; (void)n_in; (void)out_size; (void)ws_size;

  const float* x5in      = (const float*)d_in[0];
  const float* conv_w    = (const float*)d_in[1];
  const float* conv_b    = (const float*)d_in[2];
  const float* ln1_g     = (const float*)d_in[3];
  const float* ln1_b     = (const float*)d_in[4];
  const float* in_proj_w = (const float*)d_in[5];
  const float* dwconv_w  = (const float*)d_in[6];
  const float* dwconv_b  = (const float*)d_in[7];
  const float* x_proj_w  = (const float*)d_in[8];
  const float* dt_proj_w = (const float*)d_in[9];
  const float* dt_proj_b = (const float*)d_in[10];
  const float* A_logs    = (const float*)d_in[11];
  const float* Ds        = (const float*)d_in[12];
  const float* onorm_g   = (const float*)d_in[13];
  const float* onorm_b   = (const float*)d_in[14];
  const float* out_proj_w= (const float*)d_in[15];
  float* out = (float*)d_out;

  // ---- workspace carve (all pointers 256B aligned) ----
  char* base = (char*)d_ws;
  size_t cur = 0;
  auto alloc = [&](size_t bytes) -> char* {
    char* p = base + cur;
    cur += (bytes + 255) & ~(size_t)255;
    return p;
  };
  // f16 weights (sub-offsets in halves)
  const size_t WHC = 0, WHIN = 262144, WHX = 1310720, WHDT = 1572864, WHOUT = 1703936;
  _Float16* Wh    = (_Float16*)alloc(2228224 * 2);
  float*    x5r   = (float*)alloc((size_t)BL * CC * 4);
  float*    xz    = (float*)alloc((size_t)BL * 2 * DI * 4);
  float*    xc    = (float*)alloc((size_t)BL * DI * 4);
  float*    P     = (float*)alloc((size_t)KD4 * BL * CPJ * 4);
  _Float16* Dth   = (_Float16*)alloc((size_t)KD4 * BL * DTR * 2);
  _Float16* DeltaH= (_Float16*)alloc((size_t)KD4 * BL * DI * 2);
  float*    ytot  = (float*)alloc((size_t)BL * DI * 4);   // also dt_proj GEMM temp
  float*    x00   = (float*)alloc((size_t)BL * CC * 4);   // also conv GEMM temp
  _Float16* fh    = (_Float16*)alloc((size_t)BL * DI * 2);// shared f16 A-operand buf
  float*    maxper= (float*)alloc((size_t)BL * 8 * 4);
  float*    svec  = (float*)alloc((size_t)BL * 4);
  float*    maskf = (float*)alloc((size_t)BL * 4);
  float*    invn  = (float*)alloc((size_t)BL * 4);
  float*    seeds = (float*)alloc((size_t)BB * CC * 4);
  float*    cor   = (float*)alloc((size_t)BL * 4);
  float*    cormap= (float*)alloc((size_t)BL * 4);
  float*    bmn   = (float*)alloc(8 * 4);
  float*    bmx   = (float*)alloc(8 * 4);
  float*    proto = (float*)alloc(CC * 4);
  float*    dssum = (float*)alloc(DI * 4);

  // ---- convert weights to f16 ----
  k_f2h<<<cdivz(262144,256),256,0,stream>>>(conv_w,    Wh+WHC,   262144);
  k_f2h<<<cdivz(1048576,256),256,0,stream>>>(in_proj_w, Wh+WHIN, 1048576);
  k_f2h<<<cdivz(262144,256),256,0,stream>>>(x_proj_w,  Wh+WHX,   262144);
  k_f2h<<<cdivz(131072,256),256,0,stream>>>(dt_proj_w, Wh+WHDT,  131072);
  k_f2h<<<cdivz(524288,256),256,0,stream>>>(out_proj_w,Wh+WHOUT, 524288);

  const size_t NE512 = (size_t)BL * CC, NE1024 = (size_t)BL * DI;

  // ---- 1x1 conv + residual (token-major) ----
  k_tok_f16<<<cdivz(NE512,256),256,0,stream>>>(x5in, fh);
  { dim3 g(BL/128, CC/64);
    k_gemm_f16<<<g,256,0,stream>>>(fh, CC, Wh+WHC, CC, x00, CC, CC); }
  k_epi_conv<<<cdivz(NE512,256),256,0,stream>>>(x00, conv_b, x5in, x5r);

  // ---- LN1 -> in_proj ----
  k_ln1<<<BL,256,0,stream>>>(x5r, ln1_g, ln1_b, fh);
  { dim3 g(BL/128, (2*DI)/64);
    k_gemm_f16<<<g,256,0,stream>>>(fh, CC, Wh+WHIN, CC, xz, 2*DI, CC); }

  // ---- depthwise conv + silu ----
  k_dwconv<<<cdivz(NE1024,256),256,0,stream>>>(xz, dwconv_w, dwconv_b, xc);
  k_f2h<<<cdivz(NE1024,256),256,0,stream>>>(xc, fh, NE1024);

  // ---- x_proj per direction (token-major; permutation folded into scan) ----
  for (int k = 0; k < KD4; ++k) {
    dim3 g(BL/128, CPJ/64);
    k_gemm_f16<<<g,256,0,stream>>>(fh, DI, Wh+WHX + (size_t)k*CPJ*DI, DI,
                                   P + (size_t)k*BL*CPJ, CPJ, DI);
  }
  k_dtcols<<<cdivz((size_t)KD4*BL*DTR,256),256,0,stream>>>(P, Dth);

  // ---- dt_proj + softplus per direction ----
  for (int k = 0; k < KD4; ++k) {
    dim3 g(BL/128, DI/64);
    k_gemm_f16<<<g,256,0,stream>>>(Dth + (size_t)k*BL*DTR, DTR,
                                   Wh+WHDT + (size_t)k*DI*DTR, DTR,
                                   ytot, DI, DTR);
    k_softplus<<<cdivz(NE1024,256),256,0,stream>>>(ytot, dt_proj_b, DeltaH, k);
  }

  // ---- selective scan (4 directions merged via atomics) ----
  k_fill<<<cdivz(NE1024,256),256,0,stream>>>(ytot, 0.f, NE1024);
  k_dssum<<<DI/256,256,0,stream>>>(Ds, dssum);
  k_scan<<<BB*KD4*(DI/32),32,0,stream>>>(xc, DeltaH, P, A_logs, ytot);

  // ---- out_norm + gate + out_proj + residual ----
  k_ln_gate<<<BL,256,0,stream>>>(ytot, xc, dssum, xz, onorm_g, onorm_b, fh);
  { dim3 g(BL/128, CC/64);
    k_gemm_f16<<<g,256,0,stream>>>(fh, DI, Wh+WHOUT, DI, x00, CC, DI); }
  k_epi_x00<<<cdivz(NE512,256),256,0,stream>>>(x00, x5r, fh);

  // ---- gram matrix with fused row-max ----
  k_fill<<<cdivz((size_t)BL*8,256),256,0,stream>>>(maxper, -1e30f, (size_t)BL*8);
  { dim3 g(BL/128, BL/64);
    k_gram_max<<<g,256,0,stream>>>(fh, maxper); }
  k_s<<<cdivz(BL,256),256,0,stream>>>(maxper, svec);
  k_mask<<<BB,256,0,stream>>>(svec, maskf);

  // ---- correlation map ----
  k_invn<<<BL,256,0,stream>>>(x5r, invn);
  k_seeds<<<cdivz((size_t)BB*CC,256),256,0,stream>>>(x5r, invn, maskf, seeds);
  k_cor<<<BL,256,0,stream>>>(x5r, seeds, invn, cor);
  k_minmax<<<BB,256,0,stream>>>(cor, bmn, bmx);
  k_cormap<<<cdivz(BL,256),256,0,stream>>>(cor, bmn, bmx, cormap);
  k_proto<<<CC,256,0,stream>>>(x5r, cormap, proto);

  // ---- outputs ----
  k_outputs<<<cdivz((size_t)BB*CC*LL,256),256,0,stream>>>(x5r, proto, cormap, maskf, out);
}
